// LeanSelfAttention_59940563583672
// MI455X (gfx1250) — matrix operand order repaired and checked by reading, compiled, not run on hardware
//
#include <hip/hip_runtime.h>
#include <hip/hip_bf16.h>

// ---------------------------------------------------------------------------
// LeanSelfAttention for gfx1250 (MI455X): bf16 WMMA GEMMs + flash attention.
// ---------------------------------------------------------------------------

typedef __attribute__((ext_vector_type(16))) __bf16 v16bf;
typedef __attribute__((ext_vector_type(8)))  float  v8f;

union Frag16 { v16bf v; uint4 q[2]; };

static constexpr int HIDDEN = 1024;
static constexpr int HEADS  = 16;
static constexpr int HDIM   = 64;
static constexpr int SEQ    = 2048;
static constexpr int BATCH  = 4;
static constexpr int ROWS   = BATCH * SEQ;   // 8192

__device__ __forceinline__ unsigned short f2bf_bits(float f) {
  union { float f; unsigned int u; } c; c.f = f;
  unsigned int u = c.u;
  u += 0x7FFFu + ((u >> 16) & 1u);            // round-to-nearest-even
  return (unsigned short)(u >> 16);
}

// ---------------------------------------------------------------------------
// LayerNorm (fp32 math) -> bf16 activations
// ---------------------------------------------------------------------------
__global__ __launch_bounds__(256) void ln_bf16_kernel(
    const float* __restrict__ x, const float* __restrict__ gamma,
    const float* __restrict__ beta, unsigned short* __restrict__ xln) {
  __shared__ float r1[256], r2[256];
  const int row = blockIdx.x, tid = threadIdx.x;
  const float* xr = x + (size_t)row * HIDDEN;
  float v[4]; float s = 0.f, s2 = 0.f;
#pragma unroll
  for (int j = 0; j < 4; ++j) {
    v[j] = xr[tid + j * 256];
    s += v[j]; s2 += v[j] * v[j];
  }
  r1[tid] = s; r2[tid] = s2;
  __syncthreads();
  for (int off = 128; off > 0; off >>= 1) {
    if (tid < off) { r1[tid] += r1[tid + off]; r2[tid] += r2[tid + off]; }
    __syncthreads();
  }
  const float mu  = r1[0] * (1.f / HIDDEN);
  const float var = r2[0] * (1.f / HIDDEN) - mu * mu;
  const float rin = rsqrtf(var + 1e-12f);
  unsigned short* orow = xln + (size_t)row * HIDDEN;
#pragma unroll
  for (int j = 0; j < 4; ++j) {
    const int c = tid + j * 256;
    orow[c] = f2bf_bits((v[j] - mu) * rin * gamma[c] + beta[c]);
  }
}

// ---------------------------------------------------------------------------
// Weight f32 [K,N] -> bf16 transposed [N,K]  (so B-fragments load contiguously)
// ---------------------------------------------------------------------------
__global__ __launch_bounds__(256) void w_transpose_bf16(
    const float* __restrict__ w, unsigned short* __restrict__ wt,
    int K, int N) {
  const size_t idx = (size_t)blockIdx.x * 256 + threadIdx.x;
  if (idx >= (size_t)K * N) return;
  const int k = (int)(idx / N), n = (int)(idx % N);
  wt[(size_t)n * K + k] = f2bf_bits(w[idx]);
}

// ---------------------------------------------------------------------------
// Register-tiled WMMA GEMM: C[M,N] = A[M,K](bf16,row) * Bt[N,K]^T (bf16) + bias
// Block = 4 waves (128 rows x 64 cols); wave = 32(M) x 64(N):
//   per 32-wide k-step: 2 A-frags + 4 B-frags -> 8 WMMAs (24 B traffic/WMMA).
// MODE 0: scatter into Q[bh,S,HD], K[bh,S,HD], Vt[bh,HD,S] (bf16)
// MODE 1: out = acc + bias + residual (f32)
// ---------------------------------------------------------------------------
template <int MODE>
__global__ __launch_bounds__(128) void wmma_gemm_bf16(
    const unsigned short* __restrict__ A,
    const unsigned short* __restrict__ Bt,
    const float* __restrict__ bias,
    const float* __restrict__ residual,
    float* __restrict__ outF,
    unsigned short* __restrict__ qws,
    unsigned short* __restrict__ kws,
    unsigned short* __restrict__ vtws,
    int M, int N, int K) {
  const int lane = threadIdx.x & 31;
  const int wave = threadIdx.x >> 5;
  const int mrow = lane & 15, hf = lane >> 4;
  const int m0 = blockIdx.y * 128 + wave * 32;
  const int n0 = blockIdx.x * 64;

  v8f acc[2][4];
#pragma unroll
  for (int r = 0; r < 2; ++r)
#pragma unroll
    for (int t = 0; t < 4; ++t) { v8f z = {}; acc[r][t] = z; }

  const unsigned short* arow0 = A + (size_t)(m0 + mrow) * K;
  const unsigned short* arow1 = arow0 + (size_t)16 * K;
  for (int k0 = 0; k0 < K; k0 += 32) {
    Frag16 af[2];                                // A 16x32: lane = row M
    af[0].q[0] = *(const uint4*)(arow0 + k0 + hf * 8);
    af[0].q[1] = *(const uint4*)(arow0 + k0 + 16 + hf * 8);
    af[1].q[0] = *(const uint4*)(arow1 + k0 + hf * 8);
    af[1].q[1] = *(const uint4*)(arow1 + k0 + 16 + hf * 8);
#pragma unroll
    for (int t = 0; t < 4; ++t) {
      const unsigned short* brow =
          Bt + (size_t)(n0 + t * 16 + mrow) * K + k0 + hf * 8;
      Frag16 bf;                                 // B 32x16: lane = col N, K hf*8..+7 and 16+hf*8..+7
      bf.q[0] = *(const uint4*)(brow);
      bf.q[1] = *(const uint4*)(brow + 16);
      acc[0][t] = __builtin_amdgcn_wmma_f32_16x16x32_bf16(
          false, af[0].v, false, bf.v, (short)0, acc[0][t], false, false);
      acc[1][t] = __builtin_amdgcn_wmma_f32_16x16x32_bf16(
          false, af[1].v, false, bf.v, (short)0, acc[1][t], false, false);
    }
  }

#pragma unroll
  for (int r = 0; r < 2; ++r) {
#pragma unroll
    for (int t = 0; t < 4; ++t) {
#pragma unroll
      for (int i = 0; i < 8; ++i) {
        const int row = m0 + r * 16 + i + hf * 8;   // C frag: vgpr i, half hf
        const int col = n0 + t * 16 + mrow;
        const float val = acc[r][t][i] + bias[col];
        if (MODE == 1) {
          outF[(size_t)row * N + col] = val + residual[(size_t)row * N + col];
        } else {
          const int part   = col >> 10;             // 0=Q 1=K 2=V
          const int within = col & 1023;
          const int h  = within >> 6;
          const int hd = within & 63;
          const int bb = row >> 11;
          const int s  = row & 2047;
          const int bh = bb * HEADS + h;
          const unsigned short bv = f2bf_bits(val);
          if (part == 0)      qws[((size_t)bh * SEQ + s) * HDIM + hd] = bv;
          else if (part == 1) kws[((size_t)bh * SEQ + s) * HDIM + hd] = bv;
          else                vtws[((size_t)bh * HDIM + hd) * SEQ + s] = bv;
        }
      }
    }
  }
}

// ---------------------------------------------------------------------------
// Flash attention: one wave per (batch, head, 16-query tile), 64-key blocks.
// Per 64-key block: 8 WMMA for scores (HD=64 contraction), online softmax in
// C-fragment layout (row reductions via 16-lane xor shuffles), P (16x64 bf16)
// routed through LDS to A-fragment layout, 8 WMMA for P*V.
// ---------------------------------------------------------------------------
__device__ __forceinline__ float redmax16(float v) {
  v = fmaxf(v, __shfl_xor(v, 1, 32));
  v = fmaxf(v, __shfl_xor(v, 2, 32));
  v = fmaxf(v, __shfl_xor(v, 4, 32));
  v = fmaxf(v, __shfl_xor(v, 8, 32));
  return v;
}
__device__ __forceinline__ float redsum16(float v) {
  v += __shfl_xor(v, 1, 32);
  v += __shfl_xor(v, 2, 32);
  v += __shfl_xor(v, 4, 32);
  v += __shfl_xor(v, 8, 32);
  return v;
}

__global__ __launch_bounds__(128) void flash_attn_kernel(
    const unsigned short* __restrict__ Q,
    const unsigned short* __restrict__ Kk,
    const unsigned short* __restrict__ Vt,
    const float* __restrict__ amask,
    unsigned short* __restrict__ attnOut) {
  __shared__ __align__(16) unsigned short ldsP[4][16][64];   // 8 KB
  const int lane = threadIdx.x & 31;
  const int wave = threadIdx.x >> 5;
  const int wid = blockIdx.x * 4 + wave;         // 0..8191
  const int qt = wid & 127;
  const int bh = wid >> 7;                       // 0..63
  const int bb = bh >> 4;
  const int h  = bh & 15;
  const int q0 = qt * 16;
  const int mrow = lane & 15, hf = lane >> 4;

  const unsigned short* Qb = Q  + (size_t)bh * SEQ * HDIM;
  const unsigned short* Kb = Kk + (size_t)bh * SEQ * HDIM;
  const unsigned short* Vb = Vt + (size_t)bh * HDIM * SEQ;
  const float* maskb = amask + (size_t)bb * SEQ * SEQ;

  Frag16 qa[2];                                  // Q tile [16,64] = 2 k-steps
  {
    const unsigned short* qrow = Qb + (size_t)(q0 + mrow) * HDIM;
#pragma unroll
    for (int ks = 0; ks < 2; ++ks) {
      qa[ks].q[0] = *(const uint4*)(qrow + ks * 32 + hf * 8);
      qa[ks].q[1] = *(const uint4*)(qrow + ks * 32 + 16 + hf * 8);
    }
  }

  float m_run[8], l_run[8];
  v8f acc[4];
#pragma unroll
  for (int i = 0; i < 8; ++i) { m_run[i] = -1e30f; l_run[i] = 0.f; }
#pragma unroll
  for (int t = 0; t < 4; ++t) { v8f z = {}; acc[t] = z; }

  for (int kb = 0; kb < SEQ / 64; ++kb) {
    const int key0 = kb * 64;
    v8f s[4];
#pragma unroll
    for (int t = 0; t < 4; ++t) { v8f z = {}; s[t] = z; }
#pragma unroll
    for (int ks = 0; ks < 2; ++ks) {             // scores: Q[16,64] x K^T[64,64]
#pragma unroll
      for (int t = 0; t < 4; ++t) {
        Frag16 kf;
        const unsigned short* kr =
            Kb + (size_t)(key0 + t * 16 + mrow) * HDIM + ks * 32 + hf * 8;
        kf.q[0] = *(const uint4*)(kr);
        kf.q[1] = *(const uint4*)(kr + 16);
        s[t] = __builtin_amdgcn_wmma_f32_16x16x32_bf16(
            false, qa[ks].v, false, kf.v, (short)0, s[t], false, false);
      }
    }
    float p[4][8], alpha[8];
#pragma unroll
    for (int i = 0; i < 8; ++i) {                // online softmax, row q0+i+hf*8
      const int qrow = q0 + i + hf * 8;
      const float* mrowp = maskb + (size_t)qrow * SEQ + key0 + mrow;
      float a[4];
#pragma unroll
      for (int t = 0; t < 4; ++t) a[t] = s[t][i] * 0.125f + mrowp[t * 16];
      const float mx =
          redmax16(fmaxf(fmaxf(a[0], a[1]), fmaxf(a[2], a[3])));
      const float mnew = fmaxf(m_run[i], mx);
      alpha[i] = __expf(m_run[i] - mnew);
      float rsl = 0.f;
#pragma unroll
      for (int t = 0; t < 4; ++t) { p[t][i] = __expf(a[t] - mnew); rsl += p[t][i]; }
      const float rs = redsum16(rsl);
      l_run[i] = l_run[i] * alpha[i] + rs;
      m_run[i] = mnew;
    }
#pragma unroll
    for (int t = 0; t < 4; ++t)
#pragma unroll
      for (int i = 0; i < 8; ++i) acc[t][i] *= alpha[i];

    __syncthreads();                             // C-layout -> A-layout via LDS
#pragma unroll
    for (int i = 0; i < 8; ++i)
#pragma unroll
      for (int t = 0; t < 4; ++t)
        ldsP[wave][i + hf * 8][t * 16 + mrow] = f2bf_bits(p[t][i]);
    __syncthreads();
    Frag16 pa[2];                                // P as A-frags, 2 k-steps of 32
#pragma unroll
    for (int ks = 0; ks < 2; ++ks) {
      pa[ks].q[0] = *(const uint4*)&ldsP[wave][mrow][ks * 32 + hf * 8];
      pa[ks].q[1] = *(const uint4*)&ldsP[wave][mrow][ks * 32 + 16 + hf * 8];
    }
#pragma unroll
    for (int t = 0; t < 4; ++t) {                // acc += P[16,64] x V[64,64]
#pragma unroll
      for (int ks = 0; ks < 2; ++ks) {
        Frag16 vf;
        const unsigned short* vr =
            Vb + (size_t)(t * 16 + mrow) * SEQ + key0 + ks * 32 + hf * 8;
        vf.q[0] = *(const uint4*)(vr);
        vf.q[1] = *(const uint4*)(vr + 16);
        acc[t] = __builtin_amdgcn_wmma_f32_16x16x32_bf16(
            false, pa[ks].v, false, vf.v, (short)0, acc[t], false, false);
      }
    }
  }

#pragma unroll
  for (int i = 0; i < 8; ++i) l_run[i] = 1.f / l_run[i];
#pragma unroll
  for (int t = 0; t < 4; ++t) {
#pragma unroll
    for (int i = 0; i < 8; ++i) {
      const int row = bb * SEQ + q0 + i + hf * 8;
      const int col = h * HDIM + t * 16 + mrow;
      attnOut[(size_t)row * HIDDEN + col] = f2bf_bits(acc[t][i] * l_run[i]);
    }
  }
}

// ---------------------------------------------------------------------------
// Launcher
// ---------------------------------------------------------------------------
extern "C" void kernel_launch(void* const* d_in, const int* in_sizes, int n_in,
                              void* d_out, int out_size, void* d_ws, size_t ws_size,
                              hipStream_t stream) {
  const float* hidden = (const float*)d_in[0];
  const float* amask  = (const float*)d_in[1];
  const float* gamma  = (const float*)d_in[2];
  const float* beta   = (const float*)d_in[3];
  const float* Wqkv   = (const float*)d_in[4];
  const float* bqkv   = (const float*)d_in[5];
  const float* Wout   = (const float*)d_in[6];
  const float* bout   = (const float*)d_in[7];
  float* out = (float*)d_out;

  char* w = (char*)d_ws;
  unsigned short* xln   = (unsigned short*)w; w += (size_t)ROWS * HIDDEN * 2;       // 16 MB
  unsigned short* wqkvT = (unsigned short*)w; w += (size_t)3 * HIDDEN * HIDDEN * 2; //  6 MB
  unsigned short* woutT = (unsigned short*)w; w += (size_t)HIDDEN * HIDDEN * 2;     //  2 MB
  unsigned short* qws   = (unsigned short*)w; w += (size_t)ROWS * HIDDEN * 2;       // 16 MB
  unsigned short* kws   = (unsigned short*)w; w += (size_t)ROWS * HIDDEN * 2;       // 16 MB
  unsigned short* vtws  = (unsigned short*)w; w += (size_t)ROWS * HIDDEN * 2;       // 16 MB
  unsigned short* attn  = (unsigned short*)w; w += (size_t)ROWS * HIDDEN * 2;       // 16 MB

  ln_bf16_kernel<<<ROWS, 256, 0, stream>>>(hidden, gamma, beta, xln);
  w_transpose_bf16<<<(HIDDEN * 3 * HIDDEN + 255) / 256, 256, 0, stream>>>(
      Wqkv, wqkvT, HIDDEN, 3 * HIDDEN);
  w_transpose_bf16<<<(HIDDEN * HIDDEN + 255) / 256, 256, 0, stream>>>(
      Wout, woutT, HIDDEN, HIDDEN);
  wmma_gemm_bf16<0><<<dim3((3 * HIDDEN) / 64, ROWS / 128), 128, 0, stream>>>(
      xln, wqkvT, bqkv, nullptr, nullptr, qws, kws, vtws, ROWS, 3 * HIDDEN, HIDDEN);
  flash_attn_kernel<<<(BATCH * HEADS * (SEQ / 16)) / 4, 128, 0, stream>>>(
      qws, kws, vtws, amask, attn);
  wmma_gemm_bf16<1><<<dim3(HIDDEN / 64, ROWS / 128), 128, 0, stream>>>(
      attn, woutT, bout, hidden, out, nullptr, nullptr, nullptr, ROWS, HIDDEN, HIDDEN);
}
